// CoordLinear_48515950576542
// MI455X (gfx1250) — matrix operand order repaired and checked by reading, not compiled, not run on hardware
//
#include <hip/hip_runtime.h>

#define NROWS 32768
#define NCH   8
#define DDIM  256
#define ODIM  512
#define PERM_LEN (NROWS + 128)       // per-chart segments padded to 16
#define NTILES (PERM_LEN / 16)       // 2056 row-tiles, fixed grid
#define LDA    264                   // LDS row stride (bf16) -> no bank conflicts

typedef __attribute__((ext_vector_type(16))) __bf16 v16bf;
typedef __attribute__((ext_vector_type(8)))  __bf16 v8bf;
typedef __attribute__((ext_vector_type(4)))  __bf16 v4bf;
typedef __attribute__((ext_vector_type(8)))  float  v8f;

__device__ __forceinline__ float sigmoidf_(float x) {
  return 1.0f / (1.0f + __expf(-x));
}

// ---------------- Pass 1: per-row argmax over charts + histogram ----------
__global__ __launch_bounds__(256) void cl_argmax_count(
    const float* __restrict__ q, int* __restrict__ cstar, int* __restrict__ counts) {
  __shared__ int h[NCH];
  if (threadIdx.x < NCH) h[threadIdx.x] = 0;
  __syncthreads();
  int n = blockIdx.x * 256 + threadIdx.x;
  const float* row = q + (size_t)n * NCH;
  float m = row[0];
  int c = 0;
#pragma unroll
  for (int i = 1; i < NCH; ++i) {
    float v = row[i];
    if (v > m) { m = v; c = i; }   // first-max semantics == jnp.argmax
  }
  cstar[n] = c;
  atomicAdd(&h[c], 1);
  __syncthreads();
  if (threadIdx.x < NCH) atomicAdd(&counts[threadIdx.x], h[threadIdx.x]);
}

// ---------------- Pass 2: 16-aligned segment bases + perm init -----------
__global__ __launch_bounds__(256) void cl_scan_init(
    const int* __restrict__ counts, int* __restrict__ bases,
    int* __restrict__ cursor, int* __restrict__ perm) {
  if (threadIdx.x == 0) {
    int acc = 0;
#pragma unroll
    for (int c = 0; c < NCH; ++c) {
      bases[c] = acc;
      cursor[c] = acc;
      acc = (acc + counts[c] + 15) & ~15;   // next segment starts 16-aligned
    }
    bases[NCH] = acc;
  }
  for (int i = threadIdx.x; i < PERM_LEN; i += 256) perm[i] = -1;
}

// ---------------- Pass 3: scatter rows into chart segments ---------------
__global__ __launch_bounds__(256) void cl_scatter(
    const int* __restrict__ cstar, int* __restrict__ cursor, int* __restrict__ perm) {
  int n = blockIdx.x * 256 + threadIdx.x;
  int c = cstar[n];
  int pos = atomicAdd(&cursor[c], 1);
  perm[pos] = n;
}

// ---------------- Pass 4: weights f32 -> bf16 (2 MB, L2-resident) --------
__global__ __launch_bounds__(256) void cl_wconv(
    const float* __restrict__ w, __bf16* __restrict__ wbf) {
  int i = blockIdx.x * 256 + threadIdx.x;      // 4 elements per thread
  float4 v = ((const float4*)w)[i];
  v4bf o;
  o[0] = (__bf16)v.x; o[1] = (__bf16)v.y; o[2] = (__bf16)v.z; o[3] = (__bf16)v.w;
  ((v4bf*)wbf)[i] = o;
}

// ---------------- Pass 5: grouped gather-GEMM via WMMA bf16 --------------
// Block = 256 threads = 8 waves. Block tile: 16 rows x 512 outputs.
// Wave w covers outputs {ot*128 + w*16 + lane%16 : ot in 0..3}.
__global__ __launch_bounds__(256) void cl_gemm(
    const float* __restrict__ coords, const float* __restrict__ biases,
    const __bf16* __restrict__ wbf, const int* __restrict__ bases,
    const int* __restrict__ perm, float* __restrict__ out) {
  __shared__ __bf16 smA[16 * LDA];
  __shared__ int smRow[16];

  const int tile  = blockIdx.x;
  const int slot0 = tile * 16;

  // Which chart owns this tile (bases are nondecreasing, segments 16-aligned)
  int chart = 0;
#pragma unroll
  for (int c = 1; c < NCH; ++c)
    if (slot0 >= bases[c]) chart = c;

  // ---- stage A: sigmoid(coords[row, chart, :]) -> bf16 in LDS ----
  {
    const int t   = threadIdx.x;
    const int row = t >> 4;                // 0..15
    const int kc  = (t & 15) << 4;         // 0,16,...,240 (16 elems each)
    const int n   = perm[slot0 + row];
    if ((t & 15) == 0) smRow[row] = n;
    v8bf* dst = (v8bf*)&smA[row * LDA + kc];
    if (n >= 0) {
      const float4* src =
          (const float4*)(coords + ((size_t)n * NCH + chart) * DDIM + kc);
#pragma unroll
      for (int h = 0; h < 2; ++h) {
        float4 v0 = src[2 * h];
        float4 v1 = src[2 * h + 1];
        v8bf o8;
        o8[0] = (__bf16)sigmoidf_(v0.x); o8[1] = (__bf16)sigmoidf_(v0.y);
        o8[2] = (__bf16)sigmoidf_(v0.z); o8[3] = (__bf16)sigmoidf_(v0.w);
        o8[4] = (__bf16)sigmoidf_(v1.x); o8[5] = (__bf16)sigmoidf_(v1.y);
        o8[6] = (__bf16)sigmoidf_(v1.z); o8[7] = (__bf16)sigmoidf_(v1.w);
        dst[h] = o8;
      }
    } else {
      v8bf z;
#pragma unroll
      for (int i = 0; i < 8; ++i) z[i] = (__bf16)0.0f;
      dst[0] = z;
      dst[1] = z;
    }
  }
  __syncthreads();

  const int lane = threadIdx.x & 31;
  const int wave = threadIdx.x >> 5;
  const int hi   = lane >> 4;              // half-wave select
  const int lo   = lane & 15;              // A row / B column / D column

  v8f acc[4];
#pragma unroll
  for (int ot = 0; ot < 4; ++ot)
#pragma unroll
    for (int r = 0; r < 8; ++r) acc[ot][r] = 0.0f;

  // K loop: 8 x v_wmma_f32_16x16x32_bf16 per output sub-tile
#pragma unroll
  for (int kb = 0; kb < DDIM / 32; ++kb) {
    // A fragment (16-bit 16x32), this code's contiguous k order, not the ISA table order:
    //   lane(m=lo, hi): K = hi*16 + {0..15}, contiguous (matches B)
    const int ka = kb * 32 + hi * 16;
    v8bf a0 = *(const v8bf*)&smA[lo * LDA + ka];
    v8bf a1 = *(const v8bf*)&smA[lo * LDA + ka + 8];
    v16bf a;
#pragma unroll
    for (int i = 0; i < 8; ++i) { a[i] = a0[i]; a[i + 8] = a1[i]; }

#pragma unroll
    for (int ot = 0; ot < 4; ++ot) {
      // B fragment (32x16): lane(col=lo, hi): K = hi*16 + {0..15}, contiguous in D
      const int o = ot * 128 + wave * 16 + lo;
      v16bf b = *(const v16bf*)(wbf + ((size_t)chart * ODIM + o) * DDIM +
                                kb * 32 + hi * 16);
      acc[ot] = __builtin_amdgcn_wmma_f32_16x16x32_bf16(
          false, a, false, b, (short)0, acc[ot], false, false);
    }
  }

  // D layout: lane(col=lo, hi): VGPR r -> row M = r + 8*hi
#pragma unroll
  for (int ot = 0; ot < 4; ++ot) {
    const int o = ot * 128 + wave * 16 + lo;
    const float bias = biases[chart * ODIM + o];
#pragma unroll
    for (int r = 0; r < 8; ++r) {
      int n = smRow[r + 8 * hi];
      if (n >= 0) out[(size_t)n * ODIM + o] = acc[ot][r] + bias;
    }
  }
}

// -------------------------------------------------------------------------
extern "C" void kernel_launch(void* const* d_in, const int* in_sizes, int n_in,
                              void* d_out, int out_size, void* d_ws, size_t ws_size,
                              hipStream_t stream) {
  (void)in_sizes; (void)n_in; (void)out_size; (void)ws_size;
  const float* q       = (const float*)d_in[0];   // [N, C]
  const float* coords  = (const float*)d_in[1];   // [N, C, D]
  const float* weights = (const float*)d_in[2];   // [C, O, D]
  const float* biases  = (const float*)d_in[3];   // [C, O]
  float* out = (float*)d_out;                     // [N, O]

  char* ws = (char*)d_ws;
  int*    cstar  = (int*)(ws);               // 131072 B
  int*    counts = (int*)(ws + 131072);      // 32 B
  int*    bases  = (int*)(ws + 131136);      // 36 B
  int*    cursor = (int*)(ws + 131200);      // 32 B
  int*    perm   = (int*)(ws + 131264);      // 131584 B
  __bf16* wbf    = (__bf16*)(ws + 262912);   // 2097152 B (~2.25 MB total)

  hipMemsetAsync(counts, 0, NCH * sizeof(int), stream);
  cl_argmax_count<<<NROWS / 256, 256, 0, stream>>>(q, cstar, counts);
  cl_scan_init<<<1, 256, 0, stream>>>(counts, bases, cursor, perm);
  cl_scatter<<<NROWS / 256, 256, 0, stream>>>(cstar, cursor, perm);
  cl_wconv<<<(NCH * ODIM * DDIM / 4) / 256, 256, 0, stream>>>(weights, wbf);
  cl_gemm<<<dim3(NTILES, 1, 1), 256, 0, stream>>>(coords, biases, wbf, bases,
                                                  perm, out);
}
